// DynamicStructureAwareAttention_86457691668727
// MI455X (gfx1250) — compile-verified, matrix operand order checked by reading, not run on hardware
//
#include <hip/hip_runtime.h>
#include <hip/hip_bf16.h>

typedef __attribute__((ext_vector_type(16))) __bf16 v16bf;
typedef __attribute__((ext_vector_type(8)))  float  v8f;

#define BD   4
#define LQ   512
#define HD   768
#define PD   64
#define NHD  12
#define DD   64
#define PNLD 72   // padded bf16 row stride for pn in LDS
#define CHROWS 256  // paths rows staged per async chunk

// ---------------------------------------------------------------- WMMA core
__device__ __forceinline__ v8f wmma_bf16(v16bf a, v16bf b, v8f c) {
  // D = A(16x32 bf16) * B(32x16 bf16) + C(16x16 f32)
  return __builtin_amdgcn_wmma_f32_16x16x32_bf16(false, a, false, b,
                                                 (short)0, c, false, false);
}

// A fragment: row = (lane&15); per half-wave the 16 bf16 cover K = kbase+half*8+{0..7}
// and kbase+16+half*8+{0..7} (two contiguous 8-element runs).
__device__ __forceinline__ v16bf load_a_f32(const float* A, int lda, int row,
                                            int kbase, int half) {
  const float* p = A + (size_t)row * lda + kbase + half * 8;
  v16bf a;
#pragma unroll
  for (int t = 0; t < 8; ++t) { a[t] = (__bf16)p[t]; a[t + 8] = (__bf16)p[t + 16]; }
  return a;
}
__device__ __forceinline__ v16bf load_a_bf16(const __bf16* A, int lda, int row,
                                             int kbase, int half) {
  const __bf16* p = A + (size_t)row * lda + kbase + half * 8;
  v16bf a;
#pragma unroll
  for (int t = 0; t < 8; ++t) { a[t] = p[t]; a[t + 8] = p[t + 16]; }
  return a;
}
// B fragment, row-major B[k][n]: col = (lane&15), K pattern mirrors A (strided loads).
__device__ __forceinline__ v16bf load_b_rm_f32(const float* Bm, int ldb, int col,
                                               int kbase, int half) {
  v16bf b;
#pragma unroll
  for (int t = 0; t < 8; ++t) {
    b[t]     = (__bf16)Bm[(size_t)(kbase + half * 8 + t) * ldb + col];
    b[t + 8] = (__bf16)Bm[(size_t)(kbase + 16 + half * 8 + t) * ldb + col];
  }
  return b;
}
__device__ __forceinline__ v16bf load_b_rm_bf16(const __bf16* Bm, int ldb, int col,
                                                int kbase, int half) {
  v16bf b;
#pragma unroll
  for (int t = 0; t < 8; ++t) {
    b[t]     = Bm[(size_t)(kbase + half * 8 + t) * ldb + col];
    b[t + 8] = Bm[(size_t)(kbase + 16 + half * 8 + t) * ldb + col];
  }
  return b;
}
// B fragment where B[k][n] = W[n][k] (W row-major [N,K]) -> contiguous in k.
__device__ __forceinline__ v16bf load_b_tr_f32(const float* W, int ldw, int nrow,
                                               int kbase, int half) {
  const float* p = W + (size_t)nrow * ldw + kbase + half * 8;
  v16bf b;
#pragma unroll
  for (int t = 0; t < 8; ++t) { b[t] = (__bf16)p[t]; b[t + 8] = (__bf16)p[t + 16]; }
  return b;
}
__device__ __forceinline__ v16bf load_b_tr_bf16(const __bf16* W, int ldw, int nrow,
                                                int kbase, int half) {
  const __bf16* p = W + (size_t)nrow * ldw + kbase + half * 8;
  v16bf b;
#pragma unroll
  for (int t = 0; t < 8; ++t) { b[t] = p[t]; b[t + 8] = p[t + 16]; }
  return b;
}

// ---------------------------------------------------------------- K1: QKV GEMM
// [2048,768] x [768,768] per z in {Q,K,V}; outputs bf16 [B,NH,L,D]; Q scaled by D^-0.5
__global__ void qkv_gemm(const float* __restrict__ nodes,
                         const float* __restrict__ Wq, const float* __restrict__ bq,
                         const float* __restrict__ Wk, const float* __restrict__ bk,
                         const float* __restrict__ Wv, const float* __restrict__ bv,
                         __bf16* __restrict__ Q, __bf16* __restrict__ K,
                         __bf16* __restrict__ V) {
  const int lane = threadIdx.x & 31, half = lane >> 4, lm = lane & 15;
  const int m0 = blockIdx.x * 16, n0 = blockIdx.y * 16, z = blockIdx.z;
  const float* W    = (z == 0) ? Wq : (z == 1) ? Wk : Wv;
  const float* bias = (z == 0) ? bq : (z == 1) ? bk : bv;
  __bf16* out       = (z == 0) ? Q  : (z == 1) ? K  : V;
  const float scale = (z == 0) ? 0.125f : 1.0f;  // D^-0.5 = 1/8
  v8f c = {};
  for (int kb = 0; kb < HD; kb += 32) {
    v16bf a = load_a_f32(nodes, HD, m0 + lm, kb, half);
    v16bf b = load_b_rm_f32(W, HD, n0 + lm, kb, half);
    c = wmma_bf16(a, b, c);
  }
  const int n = n0 + lm, h = n >> 6, d = n & 63;
#pragma unroll
  for (int i = 0; i < 8; ++i) {
    int m = m0 + i + half * 8;
    int bb = m >> 9, l = m & 511;
    float v = (c[i] + bias[n]) * scale;
    out[(((size_t)bb * NHD + h) * LQ + l) * DD + d] = (__bf16)v;
  }
}

// ---------------------------------------------------------------- K2: qs = Q @ Wsk^T
__global__ void qs_gemm(const __bf16* __restrict__ Q, const float* __restrict__ Wsk,
                        __bf16* __restrict__ qs) {
  const int lane = threadIdx.x & 31, half = lane >> 4, lm = lane & 15;
  const int m0 = blockIdx.x * 16, n0 = blockIdx.y * 16;
  v8f c = {};
  for (int kb = 0; kb < DD; kb += 32) {
    v16bf a = load_a_bf16(Q, DD, m0 + lm, kb, half);
    v16bf b = load_b_tr_f32(Wsk, DD, n0 + lm, kb, half);  // B[d][p] = Wsk[p][d]
    c = wmma_bf16(a, b, c);
  }
#pragma unroll
  for (int i = 0; i < 8; ++i) {
    int m = m0 + i + half * 8;
    qs[(size_t)m * PD + n0 + lm] = (__bf16)c[i];
  }
}

// ---------------------------------------------------------------- K3: fused attention
// One workgroup per (q, b). 256 threads = 8 waves. Dynamic LDS layout below.
// The 256MB `paths` stream (the only HBM-bound tensor) is staged to LDS with
// async global->LDS copies (ASYNCcnt) in two 64KB chunks per row.
__global__ void attn_fused(const float* __restrict__ paths,
                           const float* __restrict__ biasT,
                           const __bf16* __restrict__ Q,
                           const __bf16* __restrict__ Kb,
                           const __bf16* __restrict__ Vb,
                           const __bf16* __restrict__ qsb,
                           const float* __restrict__ Wsv,
                           const float* __restrict__ bsv,
                           const float* __restrict__ plg,
                           const float* __restrict__ plb,
                           float* __restrict__ attn) {
  extern __shared__ char smem[];
  float*  stage = (float*)smem;            // [CHROWS][64] raw paths chunk (async dst)
  float*  S     = stage + CHROWS * PD;     // [16][512] scores
  float*  wp    = S + 16 * LQ;             // [16][64]
  float*  outac = wp + 16 * DD;            // [16][64]
  float*  brow  = outac + 16 * DD;         // [512]
  float*  lg    = brow + LQ;               // [64]
  float*  lb    = lg + PD;                 // [64]
  __bf16* pn    = (__bf16*)(lb + PD);      // [512][72]
  __bf16* qall  = pn + LQ * PNLD;          // [16][64]
  __bf16* qsall = qall + 16 * DD;          // [16][64]

  const int tid = threadIdx.x, lane = tid & 31, wid = tid >> 5;
  const int half = lane >> 4, lm = lane & 15;
  const int qi = blockIdx.x, b = blockIdx.y;

  // ---- stage small operands
  for (int i = tid; i < PD; i += 256) { lg[i] = plg[i]; lb[i] = plb[i]; }
  for (int i = tid; i < LQ; i += 256) brow[i] = biasT[(size_t)b * LQ + i];
  for (int i = tid; i < 16 * DD; i += 256) {
    int h = i >> 6, d = i & 63;
    __bf16 qv = (__bf16)0.0f, qsv = (__bf16)0.0f;
    if (h < NHD) {
      size_t off = (((size_t)b * NHD + h) * LQ + qi) * DD + d;
      qv = Q[off]; qsv = qsb[off];
    }
    qall[i] = qv; qsall[i] = qsv;
  }
  __syncthreads();

  // ---- async-stage paths[b,qi,:,:] to LDS, then LayerNorm -> pn (bf16)
  const float* prow = paths + (((size_t)b * LQ + qi) * LQ) * PD;
  const unsigned stageBase = (unsigned)(uintptr_t)stage;  // generic LDS addr低32 = LDS offset
  for (int ch = 0; ch < LQ / CHROWS; ++ch) {
    const float* gbase = prow + (size_t)ch * CHROWS * PD;
    // CHROWS*64 floats = CHROWS*16 x 16B segments; each lane copies 16B per issue.
#pragma unroll 4
    for (int seg = tid; seg < CHROWS * 16; seg += 256) {
      unsigned voff = (unsigned)seg * 16u;        // global byte offset
      unsigned loff = stageBase + voff;           // LDS byte offset
      asm volatile("global_load_async_to_lds_b128 %0, %1, %2"
                   :: "v"(loff), "v"(voff), "s"(gbase) : "memory");
    }
    asm volatile("s_wait_asynccnt 0" ::: "memory");
    __syncthreads();

    {  // one row per thread
      const int k = ch * CHROWS + tid;
      const float4* p4 = (const float4*)(stage + (size_t)tid * PD);
      float x[PD];
      float s = 0.f;
#pragma unroll
      for (int t = 0; t < 16; ++t) {
        float4 f = p4[t];
        x[4 * t] = f.x; x[4 * t + 1] = f.y; x[4 * t + 2] = f.z; x[4 * t + 3] = f.w;
        s += f.x + f.y + f.z + f.w;
      }
      float mean = s * (1.f / PD);
      float vs = 0.f;
#pragma unroll
      for (int j = 0; j < PD; ++j) { float d0 = x[j] - mean; vs += d0 * d0; }
      float inv = rsqrtf(vs * (1.f / PD) + 1e-5f);
      __bf16* dst = pn + (size_t)k * PNLD;
#pragma unroll
      for (int j = 0; j < PD; ++j)
        dst[j] = (__bf16)((x[j] - mean) * inv * lg[j] + lb[j]);
    }
    __syncthreads();  // stage reusable next chunk
  }

  // ---- S[h][k] = q_h . k_{h,k} + bias[k]   (per-head GEMV, K is L2-resident)
  for (int idx = tid; idx < 16 * LQ; idx += 256) {
    int h = idx >> 9, k = idx & 511;
    float acc = 0.f;
    if (h < NHD) {
      const __bf16* kp = Kb + (((size_t)b * NHD + h) * LQ + k) * DD;
      const __bf16* qp = qall + h * DD;
#pragma unroll 16
      for (int d = 0; d < DD; ++d) acc += (float)qp[d] * (float)kp[d];
    }
    S[idx] = acc + brow[k];
  }
  __syncthreads();

  // ---- structural scores: S[16h][512k] += qs[16h][64p] * pn^T[64p][512k] (WMMA)
  for (int t = wid; t < LQ / 16; t += 8) {
    int k0 = t * 16;
    v8f c;
#pragma unroll
    for (int i = 0; i < 8; ++i) c[i] = S[(i + half * 8) * LQ + k0 + lm];
    {
      v16bf a = load_a_bf16(qsall, DD, lm, 0, half);
      v16bf bb = load_b_tr_bf16(pn, PNLD, k0 + lm, 0, half);  // B[p][k]=pn[k][p]
      c = wmma_bf16(a, bb, c);
    }
    {
      v16bf a = load_a_bf16(qsall, DD, lm, 32, half);
      v16bf bb = load_b_tr_bf16(pn, PNLD, k0 + lm, 32, half);
      c = wmma_bf16(a, bb, c);
    }
#pragma unroll
    for (int i = 0; i < 8; ++i) S[(i + half * 8) * LQ + k0 + lm] = c[i];
  }
  __syncthreads();

  // ---- softmax over k per head row (wave handles rows wid, wid+8)
  for (int r = wid; r < 16; r += 8) {
    float* Sr = S + (size_t)r * LQ;
    float vals[16];
    float mx = -3.0e38f;
#pragma unroll
    for (int i = 0; i < 16; ++i) { vals[i] = Sr[lane + 32 * i]; mx = fmaxf(mx, vals[i]); }
#pragma unroll
    for (int off = 16; off >= 1; off >>= 1) mx = fmaxf(mx, __shfl_xor(mx, off, 32));
    float sum = 0.f;
#pragma unroll
    for (int i = 0; i < 16; ++i) { vals[i] = __expf(vals[i] - mx); sum += vals[i]; }
#pragma unroll
    for (int off = 16; off >= 1; off >>= 1) sum += __shfl_xor(sum, off, 32);
    float inv = 1.f / sum;
#pragma unroll
    for (int i = 0; i < 16; ++i) Sr[lane + 32 * i] = vals[i] * inv;
  }
  __syncthreads();

  // ---- wp[16h][64p] = w[16h][512k] * pn[512k][64p]  (WMMA, waves 0..3)
  if (wid < 4) {
    int n0 = wid * 16;
    v8f c = {};
    for (int kb = 0; kb < LQ; kb += 32) {
      v16bf a = load_a_f32(S, LQ, lm, kb, half);
      v16bf bb = load_b_rm_bf16(pn, PNLD, n0 + lm, kb, half);
      c = wmma_bf16(a, bb, c);
    }
#pragma unroll
    for (int i = 0; i < 8; ++i) wp[(i + half * 8) * DD + n0 + lm] = c[i];
  }
  // ---- out_attn[h][d] = sum_k w[h][k] * V[h][k][d]  (per-head GEMV, all waves)
  for (int idx = tid; idx < NHD * DD; idx += 256) {
    int h = idx >> 6, d = idx & 63;
    const __bf16* vp = Vb + (((size_t)b * NHD + h) * LQ) * DD + d;
    const float* Sr = S + (size_t)h * LQ;
    float acc = 0.f;
#pragma unroll 8
    for (int k = 0; k < LQ; ++k) acc += Sr[k] * (float)vp[(size_t)k * DD];
    outac[idx] = acc;
  }
  __syncthreads();

  // ---- combine: attn = out_attn + wp @ Wsv + bsv   ([B,L,H] f32)
  for (int idx = tid; idx < NHD * DD; idx += 256) {
    int h = idx >> 6, d = idx & 63;
    const float* wr = wp + (size_t)h * DD;
    float acc = outac[idx] + bsv[d];
#pragma unroll 16
    for (int p = 0; p < PD; ++p) acc += wr[p] * Wsv[(size_t)p * DD + d];
    attn[(((size_t)b * LQ) + qi) * HD + h * DD + d] = acc;
  }
}

// ---------------------------------------------------------------- K4: out proj + ReLU
__global__ void outproj(const float* __restrict__ A, const float* __restrict__ Wo,
                        const float* __restrict__ bo, float* __restrict__ Y) {
  const int lane = threadIdx.x & 31, half = lane >> 4, lm = lane & 15;
  const int m0 = blockIdx.x * 16, n0 = blockIdx.y * 16;
  v8f c = {};
  for (int kb = 0; kb < HD; kb += 32) {
    v16bf a = load_a_f32(A, HD, m0 + lm, kb, half);
    v16bf b = load_b_rm_f32(Wo, HD, n0 + lm, kb, half);
    c = wmma_bf16(a, b, c);
  }
  const int n = n0 + lm;
#pragma unroll
  for (int i = 0; i < 8; ++i) {
    int m = m0 + i + half * 8;
    Y[(size_t)m * HD + n] = fmaxf(c[i] + bo[n], 0.f);
  }
}

// ---------------------------------------------------------------- K5: residual + LN
__global__ void resid_ln(const float* __restrict__ nodes, const float* __restrict__ Y,
                         const float* __restrict__ g, const float* __restrict__ bta,
                         float* __restrict__ out) {
  __shared__ float s1[256], s2[256];
  const int m = blockIdx.x, tid = threadIdx.x;
  const float* xr = nodes + (size_t)m * HD;
  const float* yr = Y + (size_t)m * HD;
  float x[3]; float s = 0.f, q = 0.f;
#pragma unroll
  for (int j = 0; j < 3; ++j) {
    int c = tid + j * 256;
    x[j] = xr[c] + yr[c]; s += x[j]; q += x[j] * x[j];
  }
  s1[tid] = s; s2[tid] = q; __syncthreads();
  for (int off = 128; off >= 1; off >>= 1) {
    if (tid < off) { s1[tid] += s1[tid + off]; s2[tid] += s2[tid + off]; }
    __syncthreads();
  }
  float mean = s1[0] * (1.f / HD);
  float var  = s2[0] * (1.f / HD) - mean * mean;
  float inv  = rsqrtf(var + 1e-5f);
#pragma unroll
  for (int j = 0; j < 3; ++j) {
    int c = tid + j * 256;
    out[(size_t)m * HD + c] = (x[j] - mean) * inv * g[c] + bta[c];
  }
}

// ---------------------------------------------------------------- launch
extern "C" void kernel_launch(void* const* d_in, const int* in_sizes, int n_in,
                              void* d_out, int out_size, void* d_ws, size_t ws_size,
                              hipStream_t stream) {
  const float* nodes = (const float*)d_in[0];
  const float* biasT = (const float*)d_in[1];
  const float* paths = (const float*)d_in[2];
  const float* Wq = (const float*)d_in[3];  const float* bq = (const float*)d_in[4];
  const float* Wk = (const float*)d_in[5];  const float* bk = (const float*)d_in[6];
  const float* Wv = (const float*)d_in[7];  const float* bv = (const float*)d_in[8];
  const float* Wsk = (const float*)d_in[9];
  const float* Wsv = (const float*)d_in[11]; const float* bsv = (const float*)d_in[12];
  const float* Wo = (const float*)d_in[13]; const float* bo = (const float*)d_in[14];
  const float* lng = (const float*)d_in[15]; const float* lnb = (const float*)d_in[16];
  const float* plg = (const float*)d_in[17]; const float* plb = (const float*)d_in[18];
  float* out = (float*)d_out;
  (void)in_sizes; (void)n_in; (void)out_size; (void)ws_size;

  char* ws = (char*)d_ws;
  size_t off = 0;
  auto carve = [&](size_t bytes) {
    void* p = ws + off;
    off = (off + bytes + 255) & ~(size_t)255;
    return p;
  };
  const size_t nBHLD = (size_t)BD * NHD * LQ * DD;          // 24576*64
  __bf16* Qb   = (__bf16*)carve(nBHLD * sizeof(__bf16));
  __bf16* Kbf  = (__bf16*)carve(nBHLD * sizeof(__bf16));
  __bf16* Vbf  = (__bf16*)carve(nBHLD * sizeof(__bf16));
  __bf16* qsb  = (__bf16*)carve(nBHLD * sizeof(__bf16));
  float*  attn = (float*)carve((size_t)BD * LQ * HD * sizeof(float));
  float*  Ybuf = (float*)carve((size_t)BD * LQ * HD * sizeof(float));

  qkv_gemm<<<dim3((BD * LQ) / 16, HD / 16, 3), 32, 0, stream>>>(
      nodes, Wq, bq, Wk, bk, Wv, bv, Qb, Kbf, Vbf);

  qs_gemm<<<dim3((BD * NHD * LQ) / 16, PD / 16), 32, 0, stream>>>(Qb, Wsk, qsb);

  const size_t smem =
      ((size_t)CHROWS * PD + 16 * LQ + 16 * DD + 16 * DD + LQ + PD + PD) * sizeof(float) +
      ((size_t)LQ * PNLD + 16 * DD + 16 * DD) * sizeof(__bf16);  // ~183 KB < 320 KB/WGP
  attn_fused<<<dim3(LQ, BD), 256, smem, stream>>>(
      paths, biasT, Qb, Kbf, Vbf, qsb, Wsv, bsv, plg, plb, attn);

  outproj<<<dim3((BD * LQ) / 16, HD / 16), 32, 0, stream>>>(attn, Wo, bo, Ybuf);

  resid_ln<<<BD * LQ, 256, 0, stream>>>(nodes, Ybuf, lng, lnb, out);
}
